// Decoder_60112362275053
// MI455X (gfx1250) — matrix operand order repaired and checked by reading, not compiled, not run on hardware
//
#include <hip/hip_runtime.h>
#include <cstdint>

// Problem dims (match reference)
#define B_ 1024
#define T_ 256
#define Z_ 32
#define H_ 256
#define D_ 256
#define X_ 64

// LDS row strides (in dwords / floats) with padding against bank conflicts
#define HSTR 132   // bf16-pair dwords per row for h / phi buffers (H/2 = 128 + pad)
#define XSTR 20    // bf16-pair dwords per row for x buffer (Z/2 = 16 + pad)
#define OSTR 68    // floats per row for mean/std buffers (X = 64 + pad)

typedef __attribute__((ext_vector_type(16))) __bf16 v16bf;
typedef __attribute__((ext_vector_type(8)))  float  v8f;

union ABOp {
  v16bf    v;
  uint32_t u[8];
  uint4    q[2];
};

__device__ __forceinline__ uint16_t f32_to_bf16(float f) {
  uint32_t u = __float_as_uint(f);
  uint32_t r = u + 0x7FFFu + ((u >> 16) & 1u);   // round-to-nearest-even
  return (uint16_t)(r >> 16);
}
__device__ __forceinline__ float bf16_to_f32(uint16_t h) {
  return __uint_as_float(((uint32_t)h) << 16);
}
__device__ __forceinline__ uint32_t pack_bf16(float lo, float hi) {
  return (uint32_t)f32_to_bf16(lo) | ((uint32_t)f32_to_bf16(hi) << 16);
}

// ---- weight fp32 -> bf16 (row-major; pairs along K form the WMMA-B dwords)
__global__ void cvt_bf16_kernel(const float* __restrict__ src,
                                uint16_t* __restrict__ dst, int n) {
  int i = blockIdx.x * blockDim.x + threadIdx.x;
  if (i < n) dst[i] = f32_to_bf16(src[i]);
}

// A-operand (16x32 bf16) from LDS, ISA layout:
//   VGPR v, lane-half lh: kpair = (v<4 ? v : v+4) + 4*lh ; M = lane&15
// kp is contiguous for v=0..3 and v=4..7 -> compiler merges to 2x ds_load_b128
__device__ __forceinline__ void load_a(ABOp& a, const uint32_t* __restrict__ buf,
                                       int mA, int stride, int kc, int lh) {
#pragma unroll
  for (int v = 0; v < 8; ++v) {
    int kp = (v < 4 ? v : v + 4) + 4 * lh;
    a.u[v] = buf[mA * stride + kc * 16 + kp];
  }
}

// B-operand (32x16 bf16) from global row-major bf16 weights W[row][K]:
//   VGPR v, lane-half lh: kpair = (v<4 ? v : v+4) + 4*lh ; N = lane&15 selects row
__device__ __forceinline__ void load_b(ABOp& b, const uint16_t* __restrict__ W,
                                       int row, int Kp, int kc, int lh) {
  const uint32_t* dw = (const uint32_t*)W;
  int idx = row * Kp + kc * 16 + 4 * lh;         // idx % 4 == 0 -> 16B aligned
  const uint4* q = (const uint4*)(dw + idx);
  b.q[0] = q[0];
  b.q[1] = q[2];
}

__device__ __forceinline__ v8f wmma_bf16(const ABOp& a, const ABOp& b, v8f c) {
  return __builtin_amdgcn_wmma_f32_16x16x32_bf16(false, a.v, false, b.v,
                                                 (short)0, c, false, false);
}

__device__ __forceinline__ float sigmoidf_(float x) {
  return 1.0f / (1.0f + __expf(-x));
}
__device__ __forceinline__ float softplusf_(float x) {
  return (x > 20.0f) ? x : log1pf(__expf(x));
}

// Persistent recurrence kernel: one block = 16 batch rows, full T loop.
// 256 threads = 8 wave32s. Wave w owns output columns [32w, 32w+32) of every
// 256-wide activation, so GRU gate coupling stays wave-local.
__global__ void __launch_bounds__(256)
vrnn_kernel(const float* __restrict__ inp, const float* __restrict__ eps,
            const uint16_t* __restrict__ pWih, const uint16_t* __restrict__ pWhh,
            const uint16_t* __restrict__ pWphi, const uint16_t* __restrict__ pWmean,
            const uint16_t* __restrict__ pWstd,
            const float* __restrict__ b_ih, const float* __restrict__ b_hh,
            const float* __restrict__ b_phi, const float* __restrict__ b_mean,
            const float* __restrict__ b_std,
            float* __restrict__ out_xs, float* __restrict__ out_mean,
            float* __restrict__ out_std) {
  __shared__ __align__(16) uint32_t hlds[16 * HSTR];  // h (bf16 pairs)
  __shared__ __align__(16) uint32_t plds[16 * HSTR];  // phi (bf16 pairs)
  __shared__ __align__(16) uint32_t xlds[16 * XSTR];  // x_t (bf16 pairs)
  __shared__ __align__(16) float    mlds[16 * OSTR];  // mean (fp32)
  __shared__ __align__(16) float    slds[16 * OSTR];  // std  (fp32)

  const int tid  = threadIdx.x;
  const int lane = tid & 31;
  const int w    = tid >> 5;       // wave id 0..7
  const int lh   = lane >> 4;      // lane half
  const int nlo  = lane & 15;      // N (or M for A-operand)
  const int b0   = blockIdx.x * 16;

  // ---- per-lane loop-invariant bias values (hoisted once; no per-t loads)
  const int colA = 32 * w + nlo;           // tt = 0 gate-local column
  const int colB = colA + 16;              // tt = 1 gate-local column
  float bir[2] = {b_ih[colA], b_ih[colB]};
  float bhr[2] = {b_hh[colA], b_hh[colB]};
  float biz[2] = {b_ih[256 + colA], b_ih[256 + colB]};
  float bhz[2] = {b_hh[256 + colA], b_hh[256 + colB]};
  float bin[2] = {b_ih[512 + colA], b_ih[512 + colB]};
  float bhn[2] = {b_hh[512 + colA], b_hh[512 + colB]};
  float bph[2] = {b_phi[colA], b_phi[colB]};
  const int   xt = w & 3;                  // decoder output tile 0..3
  const int   xc = xt * 16 + nlo;          // decoder output column
  const float bdc = (w < 4) ? b_mean[xc] : b_std[xc];

  // h0 = 0
  for (int i = tid; i < 16 * HSTR; i += 256) hlds[i] = 0;

  for (int t = 0; t < T_; ++t) {
    // ---- launder weight bases so LLVM cannot prove the weight loads are
    // loop-invariant. Without this it hoists ~1.7KB/lane of weight tiles out
    // of the t-loop and spills them to scratch (observed in round 1). With
    // it, loads stay in-loop as global_load_b128 hitting the L2-resident
    // packed weights, and register pressure stays low.
    uintptr_t lz = 0;
    asm volatile("" : "+s"(lz));
    const uint16_t* wih  = pWih + lz;
    const uint16_t* whh  = pWhh + lz;
    const uint16_t* wphi = pWphi + lz;
    const uint16_t* wdec = ((w < 4) ? pWmean : pWstd) + lz;

    // ---- stage x_t into LDS as bf16 pairs (256 threads cover 16x16 dwords)
    {
      int m  = tid >> 4;
      int kk = tid & 15;
      const float* xp = inp + ((size_t)(b0 + m) * T_ + t) * (size_t)Z_ + kk * 2;
      float2 xv = *(const float2*)xp;
      xlds[m * XSTR + kk] = pack_bf16(xv.x, xv.y);
      if (t + 1 < T_) {  // cover serial-chain latency: prefetch next step
        __builtin_prefetch(xp + Z_, 0, 1);
        __builtin_prefetch(eps + ((size_t)(b0 + m) * T_ + (t + 1)) * (size_t)X_ + kk * 4, 0, 1);
      }
    }
    __syncthreads();

    // ---- gi = x_t @ W_ih^T  (K=32, 1 WMMA/tile), gh = h @ W_hh^T (K=256)
    v8f gia[3][2], gha[3][2];
    {
      ABOp ax;
      load_a(ax, xlds, nlo, XSTR, 0, lh);
#pragma unroll
      for (int g = 0; g < 3; ++g) {
#pragma unroll
        for (int tt = 0; tt < 2; ++tt) {
          int row = (g * 16 + 2 * w + tt) * 16 + nlo;   // output neuron
          ABOp b;
          load_b(b, wih, row, Z_ / 2, 0, lh);
          v8f z = {0.f, 0.f, 0.f, 0.f, 0.f, 0.f, 0.f, 0.f};
          gia[g][tt] = wmma_bf16(ax, b, z);
          gha[g][tt] = z;
        }
      }
#pragma unroll
      for (int kc = 0; kc < 8; ++kc) {
        ABOp ah;
        load_a(ah, hlds, nlo, HSTR, kc, lh);            // reuse A across 6 tiles
#pragma unroll
        for (int g = 0; g < 3; ++g) {
#pragma unroll
          for (int tt = 0; tt < 2; ++tt) {
            int row = (g * 16 + 2 * w + tt) * 16 + nlo;
            ABOp b;
            load_b(b, whh, row, H_ / 2, kc, lh);
            gha[g][tt] = wmma_bf16(ah, b, gha[g][tt]);
          }
        }
      }
    }
    __syncthreads();   // all waves finished reading h before h_new overwrites it

    // ---- GRU elementwise: C-layout element (r, lane) = (M=r+8*lh, N=lane&15)
    {
      uint16_t* hh = (uint16_t*)hlds;
#pragma unroll
      for (int tt = 0; tt < 2; ++tt) {
        int col = 32 * w + tt * 16 + nlo;               // 0..255 gate-local col
#pragma unroll
        for (int r = 0; r < 8; ++r) {
          int m = r + 8 * lh;
          float rg = sigmoidf_((gia[0][tt][r] + bir[tt]) + (gha[0][tt][r] + bhr[tt]));
          float zg = sigmoidf_((gia[1][tt][r] + biz[tt]) + (gha[1][tt][r] + bhz[tt]));
          float ng = tanhf((gia[2][tt][r] + bin[tt]) + rg * (gha[2][tt][r] + bhn[tt]));
          float ho = bf16_to_f32(hh[m * (HSTR * 2) + col]);
          float hn = (1.0f - zg) * ng + zg * ho;
          hh[m * (HSTR * 2) + col] = f32_to_bf16(hn);
        }
      }
    }
    __syncthreads();   // h_new complete

    // ---- phi = relu(h_new @ W_phi^T + b_phi)  (K=256)
    {
      v8f pacc[2];
#pragma unroll
      for (int tt = 0; tt < 2; ++tt) {
        v8f z = {0.f, 0.f, 0.f, 0.f, 0.f, 0.f, 0.f, 0.f};
        pacc[tt] = z;
      }
#pragma unroll
      for (int kc = 0; kc < 8; ++kc) {
        ABOp a;
        load_a(a, hlds, nlo, HSTR, kc, lh);
#pragma unroll
        for (int tt = 0; tt < 2; ++tt) {
          int row = (2 * w + tt) * 16 + nlo;
          ABOp b;
          load_b(b, wphi, row, H_ / 2, kc, lh);
          pacc[tt] = wmma_bf16(a, b, pacc[tt]);
        }
      }
      uint16_t* pp = (uint16_t*)plds;
#pragma unroll
      for (int tt = 0; tt < 2; ++tt) {
        int col = 32 * w + tt * 16 + nlo;
#pragma unroll
        for (int r = 0; r < 8; ++r) {
          int m = r + 8 * lh;
          float v = pacc[tt][r] + bph[tt];
          v = v > 0.0f ? v : 0.0f;
          pp[m * (HSTR * 2) + col] = f32_to_bf16(v);
        }
      }
    }
    __syncthreads();   // phi complete

    // ---- decoder heads. No wave-divergent WMMA: every wave runs the same
    //      loop; only operand pointers / final activation differ per wave.
    {
      float* ob = (w < 4) ? mlds : slds;
      v8f dacc = {0.f, 0.f, 0.f, 0.f, 0.f, 0.f, 0.f, 0.f};
#pragma unroll
      for (int kc = 0; kc < 8; ++kc) {
        ABOp a, b;
        load_a(a, plds, nlo, HSTR, kc, lh);
        load_b(b, wdec, xt * 16 + nlo, D_ / 2, kc, lh);
        dacc = wmma_bf16(a, b, dacc);
      }
#pragma unroll
      for (int r = 0; r < 8; ++r) {
        int m = r + 8 * lh;
        float v  = dacc[r] + bdc;
        float sg = sigmoidf_(v);
        float sp = softplusf_(v);
        ob[m * OSTR + xc] = (w < 4) ? sg : sp;
      }
    }
    __syncthreads();   // mean/std complete

    // ---- reparameterized sample + outputs (16x64 elems, 4 per thread, f4)
    {
      int e  = tid * 4;
      int m  = e >> 6;
      int x0 = e & 63;
      size_t o = ((size_t)(b0 + m) * T_ + t) * (size_t)X_ + x0;
      float4 ev = *(const float4*)(eps + o);
      float4 mv = *(const float4*)(mlds + m * OSTR + x0);
      float4 sv = *(const float4*)(slds + m * OSTR + x0);
      float4 xs;
      xs.x = ev.x * sv.x + mv.x;
      xs.y = ev.y * sv.y + mv.y;
      xs.z = ev.z * sv.z + mv.z;
      xs.w = ev.w * sv.w + mv.w;
      *(float4*)(out_xs + o)   = xs;
      *(float4*)(out_mean + o) = mv;
      *(float4*)(out_std + o)  = sv;
    }
    __syncthreads();   // protect xlds/mlds/slds before next iteration
  }
}

extern "C" void kernel_launch(void* const* d_in, const int* in_sizes, int n_in,
                              void* d_out, int out_size, void* d_ws, size_t ws_size,
                              hipStream_t stream) {
  const float* inp    = (const float*)d_in[0];
  const float* eps    = (const float*)d_in[1];
  const float* W_ih   = (const float*)d_in[2];
  const float* W_hh   = (const float*)d_in[3];
  const float* b_ih   = (const float*)d_in[4];
  const float* b_hh   = (const float*)d_in[5];
  const float* W_phi  = (const float*)d_in[6];
  const float* b_phi  = (const float*)d_in[7];
  const float* W_mean = (const float*)d_in[8];
  const float* b_mean = (const float*)d_in[9];
  const float* W_std  = (const float*)d_in[10];
  const float* b_std  = (const float*)d_in[11];

  // Workspace: bf16 weight copies (L2-resident, reused 64 blocks x 256 steps)
  uint8_t* ws = (uint8_t*)d_ws;
  uint16_t* pWih   = (uint16_t*)(ws + 0);                         //  48 KB
  uint16_t* pWhh   = (uint16_t*)(ws + 49152);                     // 384 KB
  uint16_t* pWphi  = (uint16_t*)(ws + 49152 + 393216);            // 128 KB
  uint16_t* pWmean = (uint16_t*)(ws + 49152 + 393216 + 131072);   //  32 KB
  uint16_t* pWstd  = (uint16_t*)(ws + 49152 + 393216 + 131072 + 32768);

  cvt_bf16_kernel<<<(3 * H_ * Z_ + 255) / 256, 256, 0, stream>>>(W_ih, pWih, 3 * H_ * Z_);
  cvt_bf16_kernel<<<(3 * H_ * H_ + 255) / 256, 256, 0, stream>>>(W_hh, pWhh, 3 * H_ * H_);
  cvt_bf16_kernel<<<(D_ * H_ + 255) / 256, 256, 0, stream>>>(W_phi, pWphi, D_ * H_);
  cvt_bf16_kernel<<<(X_ * D_ + 255) / 256, 256, 0, stream>>>(W_mean, pWmean, X_ * D_);
  cvt_bf16_kernel<<<(X_ * D_ + 255) / 256, 256, 0, stream>>>(W_std, pWstd, X_ * D_);

  float* out_xs   = (float*)d_out;
  float* out_mean = out_xs + (size_t)B_ * T_ * X_;
  float* out_std  = out_mean + (size_t)B_ * T_ * X_;

  vrnn_kernel<<<B_ / 16, 256, 0, stream>>>(inp, eps, pWih, pWhh, pWphi, pWmean,
                                           pWstd, b_ih, b_hh, b_phi, b_mean,
                                           b_std, out_xs, out_mean, out_std);
}
